// RouteSafetyGNN_22067541967001
// MI455X (gfx1250) — compile-verified, matrix operand order checked by reading, not run on hardware
//
#include <hip/hip_runtime.h>
#include <hip/hip_bf16.h>
#include <math.h>

typedef __attribute__((ext_vector_type(16))) __bf16 v16bf;
typedef __attribute__((ext_vector_type(8)))  __bf16 v8bf;
typedef __attribute__((ext_vector_type(8)))  float  v8f;

#define NN 100000
#define NE 3200000
#define NEL (NE + NN)

// K index for 16-bit WMMA A/B fragment element e (0..15), hi = lane>>4.
// VGPR0..3 hold K={0..7}+8*hi, VGPR4..7 hold K={16..23}+8*hi (per ISA 16-bit A layout).
__device__ __forceinline__ int kmap(int e, int hi) {
    return e + 8 * hi + (e >= 8 ? 8 : 0);
}
__device__ __forceinline__ float lrelu(float x) { return x > 0.f ? x : 0.2f * x; }

// Build a v16bf A-fragment from 32 consecutive bf16 values at rp (16B-aligned groups):
// elements 0..7 <- rp[8*hi .. 8*hi+7], elements 8..15 <- rp[16+8*hi .. 16+8*hi+7].
__device__ __forceinline__ v16bf load_a_chunk(const __bf16* rp, int hi) {
    v8bf g0 = *(const v8bf*)(rp + 8 * hi);
    v8bf g1 = *(const v8bf*)(rp + 16 + 8 * hi);
    v16bf a;
#pragma unroll
    for (int e = 0; e < 8; ++e) { a[e] = g0[e]; a[e + 8] = g1[e]; }
    return a;
}

__device__ __forceinline__ void atomicMaxF(float* addr, float val) {
    unsigned* ua = (unsigned*)addr;
    unsigned cur = *ua;
    while (__uint_as_float(cur) < val) {
        unsigned prev = atomicCAS(ua, cur, __float_as_uint(val));
        if (prev == cur) break;
        cur = prev;
    }
}

// ---------------- init workspace ----------------
__global__ void k_init(float* m1, float* s1, float* out1, float* m2, float* s2, float* emb) {
    int i = blockIdx.x * blockDim.x + threadIdx.x;
    if (i < NN * 64) out1[i] = 0.f;
    if (i < NN * 32) emb[i] = 0.f;
    if (i < NN * 2) { m1[i] = -INFINITY; s1[i] = 0.f; }
    if (i < NN)     { m2[i] = -INFINITY; s2[i] = 0.f; }
}

// ---------------- one-time weight swizzle into WMMA B-fragment order (bf16) ----------------
// w2f : 4 frags (c*2+t) x 32 lanes x 16 elems   (W2  [64,32], K chunks of 32, N tiles of 16)
// wm1f: 6 frags (c*2+t) x 32 lanes x 16 elems   (Wm1 [68,32] zero-padded to K=96)
// wm2f: 1 frag          x 32 lanes x 16 elems   (Wm2 [32,16])
__global__ void k_prepw(const float* __restrict__ W2, const float* __restrict__ Wm1,
                        const float* __restrict__ Wm2,
                        __bf16* __restrict__ w2f, __bf16* __restrict__ wm1f,
                        __bf16* __restrict__ wm2f) {
    int tid = blockIdx.x * blockDim.x + threadIdx.x;
    if (tid >= 2048 + 3072 + 512) return;
    if (tid < 2048) {
        int e = tid & 15, lane = (tid >> 4) & 31, f = tid >> 9;
        int l = lane & 15, hi = lane >> 4;
        int c = f >> 1, t = f & 1;
        int k = c * 32 + kmap(e, hi);
        w2f[tid] = (__bf16)W2[k * 32 + t * 16 + l];
    } else if (tid < 5120) {
        int idx = tid - 2048;
        int e = idx & 15, lane = (idx >> 4) & 31, f = idx >> 9;
        int l = lane & 15, hi = lane >> 4;
        int c = f >> 1, t = f & 1;
        int k = c * 32 + kmap(e, hi);
        wm1f[idx] = (__bf16)((k < 68) ? Wm1[k * 32 + t * 16 + l] : 0.0f);
    } else {
        int idx = tid - 5120;
        int e = idx & 15, lane = idx >> 4;
        int l = lane & 15, hi = lane >> 4;
        wm2f[idx] = (__bf16)Wm2[kmap(e, hi) * 16 + l];
    }
}

// ---------------- layer 1: h1 = x * W1 (outer product), attention dots ----------------
__global__ void k_node1(const float* __restrict__ x, const float* __restrict__ W1,
                        const float* __restrict__ aw_s, const float* __restrict__ aw_d,
                        float* __restrict__ h1, float* __restrict__ asrc1, float* __restrict__ adst1) {
    int n = blockIdx.x * blockDim.x + threadIdx.x;
    if (n >= NN) return;
    float xv = x[n];
    float as[2] = {0.f, 0.f}, ad[2] = {0.f, 0.f};
#pragma unroll
    for (int j = 0; j < 64; ++j) {
        float h = xv * W1[j];
        h1[n * 64 + j] = h;
        int hd = j >> 5, c = j & 31;
        as[hd] += h * aw_s[hd * 32 + c];
        ad[hd] += h * aw_d[hd * 32 + c];
    }
    asrc1[n * 2] = as[0]; asrc1[n * 2 + 1] = as[1];
    adst1[n * 2] = ad[0]; adst1[n * 2 + 1] = ad[1];
}

// ---------------- layer 1 edge softmax (3 passes) ----------------
__global__ void k_emax1(const int* __restrict__ srcL, const int* __restrict__ dstL,
                        const float* __restrict__ asrc1, const float* __restrict__ adst1,
                        float* __restrict__ m1) {
    int i = blockIdx.x * blockDim.x + threadIdx.x;
    if (i >= NEL) return;
    int s = (i < NE) ? srcL[i] : (i - NE);
    int d = (i < NE) ? dstL[i] : (i - NE);
#pragma unroll
    for (int h = 0; h < 2; ++h) {
        float ev = lrelu(asrc1[s * 2 + h] + adst1[d * 2 + h]);
        atomicMaxF(&m1[d * 2 + h], ev);
    }
}

__global__ void k_esum1(const int* __restrict__ srcL, const int* __restrict__ dstL,
                        const float* __restrict__ asrc1, const float* __restrict__ adst1,
                        const float* __restrict__ m1, float* __restrict__ s1) {
    int i = blockIdx.x * blockDim.x + threadIdx.x;
    if (i >= NEL) return;
    int s = (i < NE) ? srcL[i] : (i - NE);
    int d = (i < NE) ? dstL[i] : (i - NE);
#pragma unroll
    for (int h = 0; h < 2; ++h) {
        float ev = lrelu(asrc1[s * 2 + h] + adst1[d * 2 + h]);
        atomicAdd(&s1[d * 2 + h], __expf(ev - m1[d * 2 + h]));
    }
}

__global__ void k_escat1(const int* __restrict__ srcL, const int* __restrict__ dstL,
                         const float* __restrict__ asrc1, const float* __restrict__ adst1,
                         const float* __restrict__ m1, const float* __restrict__ s1,
                         const float* __restrict__ h1, float* __restrict__ out1) {
    long long t = (long long)blockIdx.x * blockDim.x + threadIdx.x;
    if (t >= (long long)NEL * 64) return;
    int i = (int)(t >> 6), c = (int)(t & 63), hd = c >> 5;
    int s = (i < NE) ? srcL[i] : (i - NE);
    int d = (i < NE) ? dstL[i] : (i - NE);
    float ev = lrelu(asrc1[s * 2 + hd] + adst1[d * 2 + hd]);
    float alpha = __expf(ev - m1[d * 2 + hd]) / (s1[d * 2 + hd] + 1e-16f);
    atomicAdd(&out1[d * 64 + c], h1[s * 64 + c] * alpha);
}

// relu(out1 + b1) -> bf16 activation buffer (feeds WMMA A fragments)
__global__ void k_relub1(const float* __restrict__ b1, const float* __restrict__ out1,
                         __bf16* __restrict__ out1h) {
    int i = blockIdx.x * blockDim.x + threadIdx.x;
    if (i >= NN * 64) return;
    out1h[i] = (__bf16)fmaxf(out1[i] + b1[i & 63], 0.f);
}

// ---------------- layer 2 node GEMM via WMMA: h2 = relu_out1[N,64] @ W2[64,32] ----------------
__global__ __launch_bounds__(256) void k_gemm2(const __bf16* __restrict__ out1h,
                                               const __bf16* __restrict__ w2f,
                                               float* __restrict__ h2) {
    int wid = threadIdx.x >> 5, lane = threadIdx.x & 31;
    int l = lane & 15, hi = lane >> 4;
    int base = (blockIdx.x * 8 + wid) * 16;

    v16bf B00 = *(const v16bf*)(w2f + 0 * 512 + lane * 16);
    v16bf B01 = *(const v16bf*)(w2f + 1 * 512 + lane * 16);
    v16bf B10 = *(const v16bf*)(w2f + 2 * 512 + lane * 16);
    v16bf B11 = *(const v16bf*)(w2f + 3 * 512 + lane * 16);

    int row = base + l; if (row >= NN) row = NN - 1;
    const __bf16* rp = out1h + row * 64;
    v16bf A0 = load_a_chunk(rp, hi);
    v16bf A1 = load_a_chunk(rp + 32, hi);

    v8f C0 = {}, C1 = {};
    C0 = __builtin_amdgcn_wmma_f32_16x16x32_bf16(false, A0, false, B00, (short)0, C0, false, false);
    C0 = __builtin_amdgcn_wmma_f32_16x16x32_bf16(false, A1, false, B10, (short)0, C0, false, false);
    C1 = __builtin_amdgcn_wmma_f32_16x16x32_bf16(false, A0, false, B01, (short)0, C1, false, false);
    C1 = __builtin_amdgcn_wmma_f32_16x16x32_bf16(false, A1, false, B11, (short)0, C1, false, false);

#pragma unroll
    for (int v = 0; v < 8; ++v) {
        int r = base + v + 8 * hi;
        if (r < NN) {
            h2[r * 32 + l]      = C0[v];
            h2[r * 32 + 16 + l] = C1[v];
        }
    }
}

__global__ void k_att2(const float* __restrict__ h2, const float* __restrict__ aw_s,
                       const float* __restrict__ aw_d, float* __restrict__ asrc2,
                       float* __restrict__ adst2) {
    int n = blockIdx.x * blockDim.x + threadIdx.x;
    if (n >= NN) return;
    float as = 0.f, ad = 0.f;
#pragma unroll
    for (int c = 0; c < 32; ++c) {
        float h = h2[n * 32 + c];
        as += h * aw_s[c];
        ad += h * aw_d[c];
    }
    asrc2[n] = as; adst2[n] = ad;
}

// ---------------- layer 2 edge softmax ----------------
__global__ void k_emax2(const int* __restrict__ srcL, const int* __restrict__ dstL,
                        const float* __restrict__ asrc2, const float* __restrict__ adst2,
                        float* __restrict__ m2) {
    int i = blockIdx.x * blockDim.x + threadIdx.x;
    if (i >= NEL) return;
    int s = (i < NE) ? srcL[i] : (i - NE);
    int d = (i < NE) ? dstL[i] : (i - NE);
    atomicMaxF(&m2[d], lrelu(asrc2[s] + adst2[d]));
}

__global__ void k_esum2(const int* __restrict__ srcL, const int* __restrict__ dstL,
                        const float* __restrict__ asrc2, const float* __restrict__ adst2,
                        const float* __restrict__ m2, float* __restrict__ s2) {
    int i = blockIdx.x * blockDim.x + threadIdx.x;
    if (i >= NEL) return;
    int s = (i < NE) ? srcL[i] : (i - NE);
    int d = (i < NE) ? dstL[i] : (i - NE);
    float ev = lrelu(asrc2[s] + adst2[d]);
    atomicAdd(&s2[d], __expf(ev - m2[d]));
}

__global__ void k_escat2(const int* __restrict__ srcL, const int* __restrict__ dstL,
                         const float* __restrict__ asrc2, const float* __restrict__ adst2,
                         const float* __restrict__ m2, const float* __restrict__ s2,
                         const float* __restrict__ h2, float* __restrict__ emb) {
    long long t = (long long)blockIdx.x * blockDim.x + threadIdx.x;
    if (t >= (long long)NEL * 32) return;
    int i = (int)(t >> 5), c = (int)(t & 31);
    int s = (i < NE) ? srcL[i] : (i - NE);
    int d = (i < NE) ? dstL[i] : (i - NE);
    float ev = lrelu(asrc2[s] + adst2[d]);
    float alpha = __expf(ev - m2[d]) / (s2[d] + 1e-16f);
    atomicAdd(&emb[d * 32 + c], h2[s * 32 + c] * alpha);
}

// node_embs + b2 -> bf16 embedding buffer (feeds edge-MLP A fragments)
__global__ void k_bias2(const float* __restrict__ b2, const float* __restrict__ emb,
                        __bf16* __restrict__ embh) {
    int i = blockIdx.x * blockDim.x + threadIdx.x;
    if (i >= NN * 32) return;
    embh[i] = (__bf16)(emb[i] + b2[i & 31]);
}

// ---------------- edge MLP: [E,68]@[68,32] -> relu -> @[32,16] -> relu -> @[16,1] -> sigmoid ----
__global__ __launch_bounds__(256) void k_mlp(const __bf16* __restrict__ embh,
                                             const int* __restrict__ srcL, const int* __restrict__ dstL,
                                             const float* __restrict__ ea,
                                             const __bf16* __restrict__ wm1f, const float* __restrict__ bm1,
                                             const __bf16* __restrict__ wm2f, const float* __restrict__ bm2,
                                             const float* __restrict__ Wm3, const float* __restrict__ bm3,
                                             float* __restrict__ out) {
    // per-wave staging: stride 36 floats for [16x32] tile (16B-aligned rows), 20 for [16x16]
    __shared__ float lds[8][16 * 36];
    int wid = threadIdx.x >> 5, lane = threadIdx.x & 31;
    int l = lane & 15, hi = lane >> 4;
    int base = (blockIdx.x * 8 + wid) * 16;

    // B fragments from pre-swizzled tables (2x b128 each, no cvt)
    v16bf B1[3][2];
#pragma unroll
    for (int c = 0; c < 3; ++c)
#pragma unroll
        for (int t = 0; t < 2; ++t)
            B1[c][t] = *(const v16bf*)(wm1f + (c * 2 + t) * 512 + lane * 16);
    v16bf B2 = *(const v16bf*)(wm2f + lane * 16);

    // A fragments: feat row = [embh[src] (k 0..31), embh[dst] (k 32..63), edge_attr (k 64..67), 0]
    int row = base + l; if (row >= NE) row = NE - 1;
    int s = srcL[row], d = dstL[row];
    v16bf A0 = load_a_chunk(embh + s * 32, hi);
    v16bf A1 = load_a_chunk(embh + d * 32, hi);
    v16bf A2c = {};
    if (hi == 0) {
        float4 eav = *(const float4*)(ea + (long long)row * 4);
        A2c[0] = (__bf16)eav.x; A2c[1] = (__bf16)eav.y;
        A2c[2] = (__bf16)eav.z; A2c[3] = (__bf16)eav.w;
    }

    v8f C0, C1;
#pragma unroll
    for (int v = 0; v < 8; ++v) { C0[v] = bm1[l]; C1[v] = bm1[16 + l]; }
    C0 = __builtin_amdgcn_wmma_f32_16x16x32_bf16(false, A0,  false, B1[0][0], (short)0, C0, false, false);
    C1 = __builtin_amdgcn_wmma_f32_16x16x32_bf16(false, A0,  false, B1[0][1], (short)0, C1, false, false);
    C0 = __builtin_amdgcn_wmma_f32_16x16x32_bf16(false, A1,  false, B1[1][0], (short)0, C0, false, false);
    C1 = __builtin_amdgcn_wmma_f32_16x16x32_bf16(false, A1,  false, B1[1][1], (short)0, C1, false, false);
    C0 = __builtin_amdgcn_wmma_f32_16x16x32_bf16(false, A2c, false, B1[2][0], (short)0, C0, false, false);
    C1 = __builtin_amdgcn_wmma_f32_16x16x32_bf16(false, A2c, false, B1[2][1], (short)0, C1, false, false);

    // relu + D-layout -> LDS [row][k] (stride 36 floats; rows 16B-aligned)
    float* L0 = &lds[wid][0];
#pragma unroll
    for (int v = 0; v < 8; ++v) {
        int r = v + 8 * hi;
        L0[r * 36 + l]      = fmaxf(C0[v], 0.0f);
        L0[r * 36 + 16 + l] = fmaxf(C1[v], 0.0f);
    }
    __syncthreads();

    // layer 2: A fragment from LDS via 4x ds_load_b128 + packed cvt
    v16bf A2;
    {
        const float4 fa = *(const float4*)&L0[l * 36 + 8 * hi];
        const float4 fb = *(const float4*)&L0[l * 36 + 8 * hi + 4];
        const float4 fc = *(const float4*)&L0[l * 36 + 16 + 8 * hi];
        const float4 fd = *(const float4*)&L0[l * 36 + 16 + 8 * hi + 4];
        A2[0]  = (__bf16)fa.x; A2[1]  = (__bf16)fa.y; A2[2]  = (__bf16)fa.z; A2[3]  = (__bf16)fa.w;
        A2[4]  = (__bf16)fb.x; A2[5]  = (__bf16)fb.y; A2[6]  = (__bf16)fb.z; A2[7]  = (__bf16)fb.w;
        A2[8]  = (__bf16)fc.x; A2[9]  = (__bf16)fc.y; A2[10] = (__bf16)fc.z; A2[11] = (__bf16)fc.w;
        A2[12] = (__bf16)fd.x; A2[13] = (__bf16)fd.y; A2[14] = (__bf16)fd.z; A2[15] = (__bf16)fd.w;
    }
    v8f D2;
#pragma unroll
    for (int v = 0; v < 8; ++v) D2[v] = bm2[l];
    D2 = __builtin_amdgcn_wmma_f32_16x16x32_bf16(false, A2, false, B2, (short)0, D2, false, false);
    __syncthreads();

    // relu + spill to LDS (stride 20 floats), then per-edge dot with Wm3 + sigmoid
#pragma unroll
    for (int v = 0; v < 8; ++v) {
        int r = v + 8 * hi;
        L0[r * 20 + l] = fmaxf(D2[v], 0.0f);
    }
    __syncthreads();

    if (lane < 16) {
        int eid = base + lane;
        if (eid < NE) {
            float acc = bm3[0];
#pragma unroll
            for (int q = 0; q < 4; ++q) {
                const float4 hv = *(const float4*)&L0[lane * 20 + q * 4];
                acc += hv.x * Wm3[q * 4] + hv.y * Wm3[q * 4 + 1]
                     + hv.z * Wm3[q * 4 + 2] + hv.w * Wm3[q * 4 + 3];
            }
            out[eid] = 1.0f / (1.0f + __expf(-acc));
        }
    }
}

extern "C" void kernel_launch(void* const* d_in, const int* in_sizes, int n_in,
                              void* d_out, int out_size, void* d_ws, size_t ws_size,
                              hipStream_t stream) {
    const float* x        = (const float*)d_in[0];
    const int*   ei       = (const int*)  d_in[1];
    const float* ea       = (const float*)d_in[2];
    const float* W1       = (const float*)d_in[3];
    const float* att_src1 = (const float*)d_in[4];
    const float* att_dst1 = (const float*)d_in[5];
    const float* b1       = (const float*)d_in[6];
    const float* W2       = (const float*)d_in[7];
    const float* att_src2 = (const float*)d_in[8];
    const float* att_dst2 = (const float*)d_in[9];
    const float* b2       = (const float*)d_in[10];
    const float* Wm1      = (const float*)d_in[11];
    const float* bm1      = (const float*)d_in[12];
    const float* Wm2      = (const float*)d_in[13];
    const float* bm2      = (const float*)d_in[14];
    const float* Wm3      = (const float*)d_in[15];
    const float* bm3      = (const float*)d_in[16];

    const int* srcL = ei;
    const int* dstL = ei + NE;
    float* scores = (float*)d_out;

    // workspace layout (float elements)
    float* W  = (float*)d_ws;
    float* h1    = W;                 // N*64 f32 (layer-1 messages; dead after k_escat1)
    float* asrc1 = h1 + NN * 64;      // N*2
    float* adst1 = asrc1 + NN * 2;    // N*2
    float* m1    = adst1 + NN * 2;    // N*2
    float* s1    = m1 + NN * 2;       // N*2
    float* out1  = s1 + NN * 2;       // N*64
    float* h2    = out1 + NN * 64;    // N*32
    float* asrc2 = h2 + NN * 32;      // N
    float* adst2 = asrc2 + NN;        // N
    float* m2    = adst2 + NN;        // N
    float* s2    = m2 + NN;           // N
    float* emb   = s2 + NN;           // N*32
    float* frags = emb + NN * 32;     // 2816 floats of bf16 fragment tables
    // bf16 activation buffers alias the dead h1 region:
    __bf16* out1h = (__bf16*)h1;              // N*64 bf16 = first N*32 floats of h1
    __bf16* embh  = (__bf16*)(h1 + NN * 32);  // N*32 bf16 = next N*16 floats of h1
    // pre-swizzled weight fragment tables (bf16): w2f[2048], wm1f[3072], wm2f[512]
    __bf16* w2f  = (__bf16*)frags;
    __bf16* wm1f = w2f + 2048;
    __bf16* wm2f = wm1f + 3072;

    dim3 blk(256);
    const int gN64  = (NN * 64 + 255) / 256;
    const int gN    = (NN + 255) / 256;
    const int gEL   = (NEL + 255) / 256;
    const int gEL64 = (int)(((long long)NEL * 64 + 255) / 256);
    const int gEL32 = (int)(((long long)NEL * 32 + 255) / 256);
    const int gN32  = (NN * 32 + 255) / 256;

    k_init  <<<gN64, blk, 0, stream>>>(m1, s1, out1, m2, s2, emb);
    k_prepw <<<22,   blk, 0, stream>>>(W2, Wm1, Wm2, w2f, wm1f, wm2f);
    k_node1 <<<gN,   blk, 0, stream>>>(x, W1, att_src1, att_dst1, h1, asrc1, adst1);
    k_emax1 <<<gEL,  blk, 0, stream>>>(srcL, dstL, asrc1, adst1, m1);
    k_esum1 <<<gEL,  blk, 0, stream>>>(srcL, dstL, asrc1, adst1, m1, s1);
    k_escat1<<<gEL64,blk, 0, stream>>>(srcL, dstL, asrc1, adst1, m1, s1, h1, out1);
    k_relub1<<<gN64, blk, 0, stream>>>(b1, out1, out1h);
    k_gemm2 <<<(NN + 127) / 128, blk, 0, stream>>>(out1h, w2f, h2);
    k_att2  <<<gN,   blk, 0, stream>>>(h2, att_src2, att_dst2, asrc2, adst2);
    k_emax2 <<<gEL,  blk, 0, stream>>>(srcL, dstL, asrc2, adst2, m2);
    k_esum2 <<<gEL,  blk, 0, stream>>>(srcL, dstL, asrc2, adst2, m2, s2);
    k_escat2<<<gEL32,blk, 0, stream>>>(srcL, dstL, asrc2, adst2, m2, s2, h2, emb);
    k_bias2 <<<gN32, blk, 0, stream>>>(b2, emb, embh);
    k_mlp   <<<(NE + 127) / 128, blk, 0, stream>>>(embh, srcL, dstL, ea,
                                                   wm1f, bm1, wm2f, bm2, Wm3, bm3, scores);
    (void)in_sizes; (void)n_in; (void)out_size; (void)ws_size;
}
